// LSTMNet_34471407518182
// MI455X (gfx1250) — matrix operand order repaired and checked by reading, not compiled, not run on hardware
//
#include <hip/hip_runtime.h>

// ---------------------------------------------------------------------------
// Fused LSTM for MI455X (gfx1250, wave32, WMMA).
//  Phase 0: convert W_hh / W_ih / W_out fp32 -> bf16 into d_ws (224 KB).
//  Phase 1: one workgroup per 16 batch rows (64 WGs, 256 thr = 8 waves).
//           Wave w owns hidden cols [16w,16w+16): 4 gate tiles (i,f,g,o).
//           W_hh/W_ih fragments live in VGPRs for all 512 steps; h is the
//           only recurrent state in LDS (bf16, 4KB), x tile staged per step.
//           Per step per wave: 24 x v_wmma_f32_16x16x32_bf16 (C=0 inline for
//           the first of each gate chain), hardware v_tanh_f32 activations.
// ---------------------------------------------------------------------------

#define SEQ   512
#define NF    64
#define HID   128
#define NG    512   // 4*HID
#define NOUT  128
#define ROWS  16    // batch rows per workgroup
#define NBATCH 1024

typedef __attribute__((ext_vector_type(16))) __bf16      v16bf;
typedef __attribute__((ext_vector_type(8)))  float        v8f;
typedef __attribute__((ext_vector_type(4)))  unsigned int v4u;
typedef __attribute__((ext_vector_type(4)))  __bf16       v4bf;

union Frag { v16bf f; struct { v4u lo; v4u hi; } u; };

// A-matrix fragment (16xK row-major tile, ld elements): lane m = lane&15,
// half = lane>>4.  VGPR v<4 hold K = half*8 + 2v..; v>=4 hold K = 16+half*8..
__device__ __forceinline__ v16bf load_frag_a(const __bf16* p0, int ld, int lane) {
  const int m = lane & 15, half = lane >> 4;
  const __bf16* p = p0 + m * ld + half * 8;
  Frag r;
  r.u.lo = *(const v4u*)(p);
  r.u.hi = *(const v4u*)(p + 16);
  return r.f;
}

// B-matrix fragment (Kx16 tile; source stored as W[n][k] row-major, ld elems):
// lane n = lane&15, half = lane>>4; e<8: K = half*8 + e; e>=8: K = 16+half*8+(e-8).
__device__ __forceinline__ v16bf load_frag_b(const __bf16* p0, int ld, int lane) {
  const int n = lane & 15, half = lane >> 4;
  const __bf16* p = p0 + n * ld + half * 8;
  Frag r;
  r.u.lo = *(const v4u*)(p);
  r.u.hi = *(const v4u*)(p + 16);
  return r.f;
}

// ---- fast transcendentals: single TRANS instructions, no IEEE div chains ---
#if __has_builtin(__builtin_amdgcn_tanhf)
__device__ __forceinline__ float tanh_fast(float v) {
  return __builtin_amdgcn_tanhf(v);                 // v_tanh_f32
}
__device__ __forceinline__ float sigmoid_fast(float v) {
  return __builtin_fmaf(0.5f, __builtin_amdgcn_tanhf(0.5f * v), 0.5f);
}
#else
__device__ __forceinline__ float tanh_fast(float v) {
  // 1 - 2/(e^{2v}+1): v_exp_f32 + v_rcp_f32 + fma, saturates at +/-1
  float e = __expf(2.0f * v);
  return __builtin_fmaf(-2.0f, __builtin_amdgcn_rcpf(e + 1.0f), 1.0f);
}
__device__ __forceinline__ float sigmoid_fast(float v) {
  return __builtin_amdgcn_rcpf(1.0f + __expf(-v));  // v_exp + v_add + v_rcp
}
#endif

// ---------------- Phase 0: weight conversion fp32 -> bf16 -------------------
__global__ void lstm_cvt_weights(const float* __restrict__ Whh,
                                 const float* __restrict__ Wih,
                                 const float* __restrict__ Wout,
                                 __bf16* __restrict__ ws) {
  const int nhh = NG * HID, nih = NG * NF, nout = NOUT * HID;
  int i = blockIdx.x * blockDim.x + threadIdx.x;
  if (i < nhh)                  ws[i] = (__bf16)Whh[i];
  else if (i < nhh + nih)       ws[i] = (__bf16)Wih[i - nhh];
  else if (i < nhh + nih + nout)ws[i] = (__bf16)Wout[i - nhh - nih];
}

// ---------------- Phase 1: fused recurrent kernel ---------------------------
__global__ __launch_bounds__(256) void lstm_fused(
    const float*  __restrict__ x,
    const float*  __restrict__ b_ih,
    const float*  __restrict__ b_hh,
    const float*  __restrict__ b_out,
    const __bf16* __restrict__ wWhh,   // [NG][HID] bf16
    const __bf16* __restrict__ wWih,   // [NG][NF]  bf16
    const __bf16* __restrict__ wWout,  // [NOUT][HID] bf16
    float* __restrict__ out)           // [NBATCH][NOUT] f32
{
  __shared__ __bf16 sH[ROWS * HID];  // recurrent hidden state, bf16
  __shared__ __bf16 sX[ROWS * NF];   // current x tile, bf16

  const int tid     = threadIdx.x;
  const int lane    = tid & 31;
  const int wave    = tid >> 5;
  const int rowbase = blockIdx.x * ROWS;
  const int hc0     = wave * 16;         // hidden columns owned by this wave
  const int nl      = lane & 15;

  // ---- load stationary weight fragments into VGPRs (once) ----
  v16bf whh[4][4];   // [gate][kblock]  K = kb*32
  v16bf wih[4][2];
  float biasv[4];
#pragma unroll
  for (int g = 0; g < 4; ++g) {
    const int gb = g * HID + hc0;        // gate-major row base in W
#pragma unroll
    for (int kb = 0; kb < 4; ++kb)
      whh[g][kb] = load_frag_b(wWhh + gb * HID + kb * 32, HID, lane);
#pragma unroll
    for (int kb = 0; kb < 2; ++kb)
      wih[g][kb] = load_frag_b(wWih + gb * NF + kb * 32, NF, lane);
    biasv[g] = b_ih[gb + nl] + b_hh[gb + nl];
  }

  // ---- init: h = 0 in LDS, stage x(t=0), c = 0 in registers ----
  for (int i = tid; i < ROWS * HID; i += 256) sH[i] = (__bf16)0.0f;
  {
    const int e0 = tid * 4;                       // 256 thr * 4 = 16*64 elems
    const int r = e0 >> 6, f = e0 & 63;
    const float4 v = *(const float4*)(x + ((size_t)(rowbase + r) * SEQ + 0) * NF + f);
    v4bf b4; b4.x = (__bf16)v.x; b4.y = (__bf16)v.y;
             b4.z = (__bf16)v.z; b4.w = (__bf16)v.w;
    *(v4bf*)(sX + e0) = b4;
  }
  float cc[8];
#pragma unroll
  for (int e = 0; e < 8; ++e) cc[e] = 0.0f;

  const v8f kzero = {};   // C = 0 (lowers to inline SRC2=0)

  // ---- recurrence over time ----
  for (int t = 0; t < SEQ; ++t) {
    __syncthreads();   // sX(t) and sH(t) visible

    const v16bf ax0 = load_frag_a(sX + 0,  NF,  lane);
    const v16bf ax1 = load_frag_a(sX + 32, NF,  lane);
    v16bf ah[4];
#pragma unroll
    for (int kb = 0; kb < 4; ++kb)
      ah[kb] = load_frag_a(sH + kb * 32, HID, lane);

    // gate pre-activations, bias folded into the activation stage.
    // Interleaved g-inner so consecutive WMMAs share A and write
    // independent accumulators.
    v8f acc[4];
#pragma unroll
    for (int g = 0; g < 4; ++g)
      acc[g] = __builtin_amdgcn_wmma_f32_16x16x32_bf16(
                   false, ax0, false, wih[g][0], (short)0, kzero, false, false);
#pragma unroll
    for (int g = 0; g < 4; ++g)
      acc[g] = __builtin_amdgcn_wmma_f32_16x16x32_bf16(
                   false, ax1, false, wih[g][1], (short)0, acc[g], false, false);
#pragma unroll
    for (int kb = 0; kb < 4; ++kb)
#pragma unroll
      for (int g = 0; g < 4; ++g)
        acc[g] = __builtin_amdgcn_wmma_f32_16x16x32_bf16(
                     false, ah[kb], false, whh[g][kb], (short)0, acc[g], false, false);

    // elementwise cell update (C/D layout: m = r + 8*(lane>>4), n = lane&15)
    float hval[8];
#pragma unroll
    for (int e = 0; e < 8; ++e) {
      const float iv = sigmoid_fast(acc[0][e] + biasv[0]);
      const float fv = sigmoid_fast(acc[1][e] + biasv[1]);
      const float gv = tanh_fast   (acc[2][e] + biasv[2]);
      const float ov = sigmoid_fast(acc[3][e] + biasv[3]);
      cc[e]   = fv * cc[e] + iv * gv;
      hval[e] = ov * tanh_fast(cc[e]);
    }

    __syncthreads();   // everyone done reading sH(t)/sX(t)

    // publish h(t+1) as bf16
#pragma unroll
    for (int r = 0; r < 8; ++r) {
      const int m = r + 8 * (lane >> 4);
      sH[m * HID + hc0 + nl] = (__bf16)hval[r];
    }
    // stage x(t+1)
    if (t + 1 < SEQ) {
      const int e0 = tid * 4;
      const int r = e0 >> 6, f = e0 & 63;
      const float* xp = x + ((size_t)(rowbase + r) * SEQ + (t + 1)) * NF + f;
      const float4 v = *(const float4*)xp;
      v4bf b4; b4.x = (__bf16)v.x; b4.y = (__bf16)v.y;
               b4.z = (__bf16)v.z; b4.w = (__bf16)v.w;
      *(v4bf*)(sX + e0) = b4;
      if (t + 2 < SEQ)   // warm L2/L0 for the step after next
        __builtin_prefetch(xp + NF, 0, 1);
    }
  }

  __syncthreads();   // final h in sH

  // ---- output projection: out = h @ W_out^T + b_out ----
  v8f oacc = __builtin_amdgcn_wmma_f32_16x16x32_bf16(
                 false, load_frag_a(sH + 0, HID, lane),
                 false, load_frag_b(wWout + hc0 * HID + 0, HID, lane),
                 (short)0, kzero, false, false);
#pragma unroll
  for (int kb = 1; kb < 4; ++kb) {
    const v16bf a = load_frag_a(sH + kb * 32, HID, lane);
    const v16bf b = load_frag_b(wWout + hc0 * HID + kb * 32, HID, lane);
    oacc = __builtin_amdgcn_wmma_f32_16x16x32_bf16(
               false, a, false, b, (short)0, oacc, false, false);
  }
  const float bo = b_out[hc0 + nl];
#pragma unroll
  for (int r = 0; r < 8; ++r) {
    const int m = r + 8 * (lane >> 4);
    out[(size_t)(rowbase + m) * NOUT + hc0 + nl] = oacc[r] + bo;
  }
}

// ---------------------------------------------------------------------------
extern "C" void kernel_launch(void* const* d_in, const int* in_sizes, int n_in,
                              void* d_out, int out_size, void* d_ws, size_t ws_size,
                              hipStream_t stream) {
  const float* x     = (const float*)d_in[0];
  const float* W_ih  = (const float*)d_in[1];
  const float* W_hh  = (const float*)d_in[2];
  const float* b_ih  = (const float*)d_in[3];
  const float* b_hh  = (const float*)d_in[4];
  const float* W_out = (const float*)d_in[5];
  const float* b_out = (const float*)d_in[6];
  float* out = (float*)d_out;

  __bf16* ws    = (__bf16*)d_ws;
  __bf16* wWhh  = ws;                       // NG*HID
  __bf16* wWih  = wWhh + NG * HID;          // NG*NF
  __bf16* wWout = wWih + NG * NF;           // NOUT*HID

  const int total = NG * HID + NG * NF + NOUT * HID;
  lstm_cvt_weights<<<(total + 255) / 256, 256, 0, stream>>>(W_hh, W_ih, W_out, ws);
  lstm_fused<<<NBATCH / ROWS, 256, 0, stream>>>(x, b_ih, b_hh, b_out,
                                                wWhh, wWih, wWout, out);
}
